// SentimentModelWithRegAttention_31275951849713
// MI455X (gfx1250) — compile-verified, matrix operand order checked by reading, not run on hardware
//
#include <hip/hip_runtime.h>
#include <hip/hip_bf16.h>
#include <math.h>

// Model dims (fixed by reference)
#define Bn 256
#define Tn 512
#define Dn 256
#define Hn 256
#define An 128
#define G4 1024   // 4*H

typedef __attribute__((ext_vector_type(16))) __bf16 v16bf;
typedef __attribute__((ext_vector_type(8)))  float  v8f;
typedef unsigned int u32x4 __attribute__((ext_vector_type(4)));
typedef int          i32x4 __attribute__((ext_vector_type(4)));
typedef int          i32x8 __attribute__((ext_vector_type(8)));

#if defined(__HIP_DEVICE_COMPILE__) && __has_builtin(__builtin_amdgcn_tensor_load_to_lds)
#define HAVE_TDM 1
#else
#define HAVE_TDM 0
#endif

__device__ __forceinline__ float sigmoidf_(float x) { return 1.0f / (1.0f + __expf(-x)); }

// Load one 16-lane-layout bf16 WMMA fragment: two contiguous 16B runs
// p0 must point at (row_base + k0 + 8*half); second run is +16 elements.
__device__ __forceinline__ v16bf frag_ld(const __bf16* p0) {
    union { v16bf v; uint4 u[2]; } f;
    f.u[0] = *(const uint4*)(p0);
    f.u[1] = *(const uint4*)(p0 + 16);
    return f.v;
}

__device__ __forceinline__ v8f wmma_bf16(v16bf a, v16bf b, v8f c) {
    return __builtin_amdgcn_wmma_f32_16x16x32_bf16(false, a, false, b, (short)0, c, false, false);
}

#if HAVE_TDM
// Issue a TDM descriptor (wave-level; EXEC ignored). Groups per CDNA5 ISA.
__device__ __forceinline__ void tdm_issue(u32x4 g0, i32x8 g1, i32x4 g2, i32x4 g3) {
#if __clang_major__ >= 23
    i32x8 z8 = {0, 0, 0, 0, 0, 0, 0, 0};
    __builtin_amdgcn_tensor_load_to_lds(g0, g1, g2, g3, z8, 0);
#else
    __builtin_amdgcn_tensor_load_to_lds(g0, g1, g2, g3, 0);
#endif
}

// 2D f32 tile load: tile_d0 x tile_d1 elements, row stride stride0 (elements).
__device__ __forceinline__ void tdm_load_2d_f32(unsigned lds_addr, const void* gptr,
                                                unsigned d0, unsigned d1,
                                                unsigned long long stride0) {
    unsigned long long ga = (unsigned long long)(uintptr_t)gptr;
    u32x4 g0;
    g0[0] = 1u;                                              // count=1
    g0[1] = lds_addr;
    g0[2] = (unsigned)(ga & 0xFFFFFFFFu);
    g0[3] = (unsigned)((ga >> 32) & 0x01FFFFFFu) | (2u << 30);  // type=2
    i32x8 g1;
    g1[0] = (2 << 16);                                       // data_size=4B
    g1[1] = (int)((d0 & 0xFFFFu) << 16);                     // tensor_dim0 lo
    g1[2] = (int)(((d0 >> 16) & 0xFFFFu) | ((d1 & 0xFFFFu) << 16));
    g1[3] = (int)(((d1 >> 16) & 0xFFFFu) | ((d0 & 0xFFFFu) << 16)); // tile_dim0 = d0
    g1[4] = (int)(d1 & 0xFFFFu);                             // tile_dim1 = d1
    g1[5] = (int)(stride0 & 0xFFFFFFFFull);
    g1[6] = (int)((stride0 >> 32) & 0xFFFFull);
    g1[7] = 0;
    i32x4 z = {0, 0, 0, 0};
    tdm_issue(g0, g1, z, z);
}

// Gather 16 rows (16-bit indices) of row_len f32 elements each into LDS.
__device__ __forceinline__ void tdm_gather16_f32(unsigned lds_addr, const void* gptr,
                                                 unsigned row_len, unsigned nrows,
                                                 const int idxw[8]) {
    unsigned long long ga = (unsigned long long)(uintptr_t)gptr;
    u32x4 g0;
    g0[0] = 1u | 0x80000000u;                                // count=1, gather_mode, 16b idx
    g0[1] = lds_addr;
    g0[2] = (unsigned)(ga & 0xFFFFFFFFu);
    g0[3] = (unsigned)((ga >> 32) & 0x01FFFFFFu) | (2u << 30);
    i32x8 g1;
    g1[0] = (2 << 16);                                       // data_size=4B
    g1[1] = (int)((row_len & 0xFFFFu) << 16);                // tensor_dim0 lo
    g1[2] = (int)(((row_len >> 16) & 0xFFFFu) | ((nrows & 0xFFFFu) << 16));
    g1[3] = (int)(((nrows >> 16) & 0xFFFFu) | ((row_len & 0xFFFFu) << 16)); // tile_dim0
    g1[4] = 16;                                              // tile_dim1 = #indices
    g1[5] = (int)row_len;                                    // stride0
    g1[6] = 0;
    g1[7] = 0;
    i32x4 g2 = {idxw[0], idxw[1], idxw[2], idxw[3]};
    i32x4 g3 = {idxw[4], idxw[5], idxw[6], idxw[7]};
    tdm_issue(g0, g1, g2, g3);
}
#endif // HAVE_TDM

__device__ __forceinline__ unsigned lds_off(const void* p) {
    // generic shared pointer: low 32 bits are the LDS byte address
    return (unsigned)(uintptr_t)p;
}

// ---------------- K0: seq_len[b] = sum_t sign(x[b,t]) ----------------
__global__ void k_seqlen(const int* __restrict__ x, int* __restrict__ seqlen) {
    __shared__ int red[256];
    int b = blockIdx.x, tid = threadIdx.x;
    int c = (x[b * Tn + tid] != 0) + (x[b * Tn + 256 + tid] != 0);
    red[tid] = c;
    __syncthreads();
    for (int s = 128; s > 0; s >>= 1) { if (tid < s) red[tid] += red[tid + s]; __syncthreads(); }
    if (tid == 0) seqlen[b] = red[0];
}

// ---------------- K1: weight prep (transpose + f32->bf16) ----------------
__global__ void k_prep(const float* __restrict__ lk, const float* __restrict__ wo,
                       __bf16* __restrict__ WxT, __bf16* __restrict__ WhT,
                       __bf16* __restrict__ WoT) {
    int i = blockIdx.x * blockDim.x + threadIdx.x;
    int stride = gridDim.x * blockDim.x;
    for (int idx = i; idx < 512 * G4; idx += stride) {
        int d = idx / G4, n = idx % G4;
        __bf16 v = (__bf16)lk[idx];
        if (d < Dn) WxT[n * Dn + d] = v;
        else        WhT[n * Hn + (d - Dn)] = v;
    }
    for (int idx = i; idx < Hn * An; idx += stride) {
        int h = idx / An, a = idx % An;
        WoT[a * Hn + h] = (__bf16)wo[idx];
    }
}

// ---------------- K2: fused embedding gather (TDM) + xw = e@Wx + bias ----------------
__global__ void __launch_bounds__(512) k_xw(const int* __restrict__ x,
                                            const float* __restrict__ embed_w,
                                            const __bf16* __restrict__ WxT,
                                            const float* __restrict__ bias,
                                            float* __restrict__ xw) {
    __shared__ __align__(16) float  Sg[16][Dn];   // gathered embedding rows (f32)
    __shared__ __align__(16) __bf16 Ab[16][Dn];   // bf16 A tile
    int tid = threadIdx.x, mtile = blockIdx.x;
#if HAVE_TDM
    if (tid < 32) {  // wave 0 issues the TDM gather of 16 embedding rows
        int idxw[8];
        const int* xb = x + mtile * 16;
        #pragma unroll
        for (int i = 0; i < 8; ++i)
            idxw[i] = (xb[2 * i] & 0xFFFF) | (xb[2 * i + 1] << 16);
        tdm_gather16_f32(lds_off(&Sg[0][0]), embed_w, Dn, 50000, idxw);
        __builtin_amdgcn_s_wait_tensorcnt(0);
    }
    __syncthreads();
    {   // convert staged rows f32 -> bf16
        int r = tid >> 5, c0 = (tid & 31) * 8;
        #pragma unroll
        for (int j = 0; j < 8; ++j) Ab[r][c0 + j] = (__bf16)Sg[r][c0 + j];
    }
#else
    {
        int r = tid >> 5, c0 = (tid & 31) * 8;
        int tok = x[mtile * 16 + r];
        const float* src = embed_w + (long)tok * Dn + c0;
        #pragma unroll
        for (int j = 0; j < 8; ++j) Ab[r][c0 + j] = (__bf16)src[j];
    }
#endif
    __syncthreads();
    int wave = tid >> 5, lane = tid & 31, m = lane & 15, half = lane >> 4;
    #pragma unroll
    for (int q = 0; q < 4; ++q) {
        int nt = wave * 4 + q;            // 0..63 over 1024 cols
        v8f acc = {};
        #pragma unroll
        for (int kk = 0; kk < 8; ++kk) {
            int k0 = kk * 32;
            v16bf a = frag_ld(&Ab[m][k0 + 8 * half]);
            v16bf b = frag_ld(WxT + (long)(nt * 16 + m) * Dn + k0 + 8 * half);
            acc = wmma_bf16(a, b, acc);
        }
        int n = nt * 16 + m;
        float bn = bias[n];
        #pragma unroll
        for (int r = 0; r < 8; ++r) {
            int row = r + 8 * half;
            xw[(long)(mtile * 16 + row) * G4 + n] = acc[r] + bn;
        }
    }
}

// ---------------- K3: recurrent LSTM scan ----------------
// 16 WGs, each owns a 16-row batch tile. h in LDS (bf16), c in registers.
// Per-step xw loads are hoisted before the GEMM to overlap with WMMA work.
__global__ void __launch_bounds__(512) k_lstm(const float* __restrict__ xw,
                                              const __bf16* __restrict__ WhT,
                                              const int* __restrict__ seqlen,
                                              float* __restrict__ outs) {
    __shared__ __align__(16) __bf16 hb[16][Hn];   // h state (bf16 for WMMA A)
    __shared__ float zb[16][G4];                  // z = h @ Wh
    __shared__ int   sl[16];
    int tid = threadIdx.x;
    int b0  = blockIdx.x * 16;
    for (int i = tid; i < 16 * Hn; i += 512) hb[i >> 8][i & 255] = (__bf16)0.0f;
    if (tid < 16) sl[tid] = seqlen[b0 + tid];
    // warm L2 with WhT (re-read every step)
    for (int i = tid; i < (G4 * Hn * 2) / 128; i += 512)
        __builtin_prefetch((const char*)WhT + i * 128, 0, 3);
    __syncthreads();
    int wave = tid >> 5, lane = tid & 31, m = lane & 15, half = lane >> 4;
    int row  = tid >> 5;             // elementwise mapping (fixed across steps)
    int c0   = (tid & 31) * 8;
    int slr  = sl[row];
    float creg[8];
    #pragma unroll
    for (int j = 0; j < 8; ++j) creg[j] = 0.0f;

    for (int t = 0; t < Tn; ++t) {
        // hoisted gate-input loads (overlap with WMMA below via LOADcnt)
        long base = ((long)(b0 + row) * Tn + t) * G4;
        float4 xg[8];
        #pragma unroll
        for (int g = 0; g < 4; ++g) {
            xg[2 * g]     = *(const float4*)(xw + base + g * 256 + c0);
            xg[2 * g + 1] = *(const float4*)(xw + base + g * 256 + c0 + 4);
        }
        // GEMM: (16 x 256) @ (256 x 1024) -> zb
        #pragma unroll
        for (int q = 0; q < 4; ++q) {
            int nt = wave * 4 + q;
            v8f acc = {};
            #pragma unroll
            for (int kk = 0; kk < 8; ++kk) {
                int k0 = kk * 32;
                v16bf a = frag_ld(&hb[m][k0 + 8 * half]);
                v16bf b = frag_ld(WhT + (long)(nt * 16 + m) * Hn + k0 + 8 * half);
                acc = wmma_bf16(a, b, acc);
            }
            int n = nt * 16 + m;
            #pragma unroll
            for (int r = 0; r < 8; ++r) zb[r + 8 * half][n] = acc[r];
        }
        __syncthreads();
        // gates + state update, 8 elements per thread
        {
            int mflag = (t < slr);
            const float* xi = (const float*)&xg[0];
            const float* xj = (const float*)&xg[2];
            const float* xf = (const float*)&xg[4];
            const float* xo = (const float*)&xg[6];
            #pragma unroll
            for (int j = 0; j < 8; ++j) {
                int col = c0 + j;
                float zi = zb[row][col]       + xi[j];
                float zj = zb[row][col + 256] + xj[j];
                float zf = zb[row][col + 512] + xf[j];
                float zo = zb[row][col + 768] + xo[j];
                float cnew = creg[j] * sigmoidf_(zf + 1.0f) + sigmoidf_(zi) * tanhf(zj);
                float hnew = tanhf(cnew) * sigmoidf_(zo);
                outs[((long)(b0 + row) * Tn + t) * Hn + col] = mflag ? hnew : 0.0f;
                if (mflag) { creg[j] = cnew; hb[row][col] = (__bf16)hnew; }
            }
        }
        __syncthreads();
    }
}

// ---------------- K4: vu[b,t] = sum_a tanh(outs@w_omega + b_omega)[a]*u_omega[a] ----------------
__global__ void __launch_bounds__(256) k_vu(const float* __restrict__ outs,
                                            const __bf16* __restrict__ WoT,
                                            const float* __restrict__ b_omega,
                                            const float* __restrict__ u_omega,
                                            float* __restrict__ vu) {
    __shared__ __align__(16) float  Sg[16][Hn];
    __shared__ __align__(16) __bf16 Ab[16][Hn];
    __shared__ float vuloc[16];
    int tid = threadIdx.x, mtile = blockIdx.x;
#if HAVE_TDM
    if (tid < 32) {  // wave 0: TDM-copy the contiguous 16x256 f32 tile into LDS
        tdm_load_2d_f32(lds_off(&Sg[0][0]), outs + (long)mtile * 16 * Hn,
                        /*d0=*/4096, /*d1=*/4, /*stride0=*/4096);
        __builtin_amdgcn_s_wait_tensorcnt(0);
    }
    __syncthreads();
    {
        int r = tid >> 4, c0 = (tid & 15) * 16;
        #pragma unroll
        for (int j = 0; j < 16; ++j) Ab[r][c0 + j] = (__bf16)Sg[r][c0 + j];
    }
#else
    {
        int r = tid >> 4, c0 = (tid & 15) * 16;
        const float* src = outs + (long)(mtile * 16 + r) * Hn + c0;
        #pragma unroll
        for (int j = 0; j < 16; ++j) Ab[r][c0 + j] = (__bf16)src[j];
    }
#endif
    if (tid < 16) vuloc[tid] = 0.0f;
    __syncthreads();
    int wave = tid >> 5, lane = tid & 31, m = lane & 15, half = lane >> 4;
    int nt = wave;  // 8 tiles over A=128
    v8f acc = {};
    #pragma unroll
    for (int kk = 0; kk < 8; ++kk) {
        int k0 = kk * 32;
        v16bf a = frag_ld(&Ab[m][k0 + 8 * half]);
        v16bf b = frag_ld(WoT + (long)(nt * 16 + m) * Hn + k0 + 8 * half);
        acc = wmma_bf16(a, b, acc);
    }
    int n = nt * 16 + m;
    float bo = b_omega[n], uo = u_omega[n];
    #pragma unroll
    for (int r = 0; r < 8; ++r) {
        float v = tanhf(acc[r] + bo) * uo;
        atomicAdd(&vuloc[r + 8 * half], v);
    }
    __syncthreads();
    if (tid < 16) vu[mtile * 16 + tid] = vuloc[tid];
}

// ---------------- K5: softmax(vu) over T, weighted-sum outs, head + softmax ----------------
__global__ void __launch_bounds__(256) k_final(const float* __restrict__ vu,
                                               const float* __restrict__ outs,
                                               const float* __restrict__ w,
                                               const float* __restrict__ b2,
                                               float* __restrict__ out) {
    __shared__ float alpha[Tn];
    __shared__ float red[256];
    int b = blockIdx.x, tid = threadIdx.x;
    float v0 = vu[b * Tn + tid], v1 = vu[b * Tn + 256 + tid];
    red[tid] = fmaxf(v0, v1); __syncthreads();
    for (int s = 128; s > 0; s >>= 1) { if (tid < s) red[tid] = fmaxf(red[tid], red[tid + s]); __syncthreads(); }
    float mx = red[0]; __syncthreads();
    float e0 = __expf(v0 - mx), e1 = __expf(v1 - mx);
    alpha[tid] = e0; alpha[tid + 256] = e1;
    red[tid] = e0 + e1; __syncthreads();
    for (int s = 128; s > 0; s >>= 1) { if (tid < s) red[tid] += red[tid + s]; __syncthreads(); }
    float denom = red[0]; __syncthreads();
    float acc = 0.0f;
    for (int t = 0; t < Tn; ++t) acc += alpha[t] * outs[((long)b * Tn + t) * Hn + tid];
    acc /= denom;
    float l0p = acc * w[tid * 2 + 0];
    float l1p = acc * w[tid * 2 + 1];
    red[tid] = l0p; __syncthreads();
    for (int s = 128; s > 0; s >>= 1) { if (tid < s) red[tid] += red[tid + s]; __syncthreads(); }
    float l0 = red[0] + b2[0]; __syncthreads();
    red[tid] = l1p; __syncthreads();
    for (int s = 128; s > 0; s >>= 1) { if (tid < s) red[tid] += red[tid + s]; __syncthreads(); }
    float l1 = red[0] + b2[1];
    if (tid == 0) {
        float mm = fmaxf(l0, l1);
        float a0 = __expf(l0 - mm), a1 = __expf(l1 - mm);
        out[b * 2 + 0] = a0 / (a0 + a1);
        out[b * 2 + 1] = a1 / (a0 + a1);
    }
}

extern "C" void kernel_launch(void* const* d_in, const int* in_sizes, int n_in,
                              void* d_out, int out_size, void* d_ws, size_t ws_size,
                              hipStream_t stream) {
    const int*   x        = (const int*)d_in[0];
    const float* embed_w  = (const float*)d_in[1];
    const float* lk       = (const float*)d_in[2];
    const float* lb       = (const float*)d_in[3];
    const float* wo       = (const float*)d_in[4];
    const float* bo       = (const float*)d_in[5];
    const float* uo       = (const float*)d_in[6];
    const float* wcls     = (const float*)d_in[7];
    const float* bcls     = (const float*)d_in[8];
    float* out = (float*)d_out;

    char* p = (char*)d_ws;
    auto alloc = [&](size_t bytes) { void* r = (void*)p; p += (bytes + 255) & ~(size_t)255; return r; };
    float*  xw     = (float*)alloc((size_t)Bn * Tn * G4 * 4);   // 537 MB
    float*  outs   = (float*)alloc((size_t)Bn * Tn * Hn * 4);   // 134 MB
    __bf16* WxT    = (__bf16*)alloc((size_t)G4 * Dn * 2);
    __bf16* WhT    = (__bf16*)alloc((size_t)G4 * Hn * 2);
    __bf16* WoT    = (__bf16*)alloc((size_t)An * Hn * 2);
    float*  vubuf  = (float*)alloc((size_t)Bn * Tn * 4);
    int*    seqlen = (int*)alloc((size_t)Bn * 4);

    k_seqlen<<<Bn, 256, 0, stream>>>(x, seqlen);
    k_prep<<<512, 256, 0, stream>>>(lk, wo, WxT, WhT, WoT);
    k_xw<<<(Bn * Tn) / 16, 512, 0, stream>>>(x, embed_w, WxT, lb, xw);
    k_lstm<<<Bn / 16, 512, 0, stream>>>(xw, WhT, seqlen, outs);
    k_vu<<<(Bn * Tn) / 16, 256, 0, stream>>>(outs, WoT, bo, uo, vubuf);
    k_final<<<Bn, 256, 0, stream>>>(vubuf, outs, wcls, bcls, out);
}
